// MultiHeadSelfAttention_57861799412229
// MI455X (gfx1250) — matrix operand order repaired and checked by reading, not compiled, not run on hardware
//
#include <hip/hip_runtime.h>

// ---------------------------------------------------------------------------
// MI455X (gfx1250) multi-head self-attention forward.
// bf16 WMMA (v_wmma_f32_16x16x32_bf16) for all four GEMMs, f32 accumulate.
// Tile staging uses CDNA5 async global->LDS copies (ASYNCcnt) + double
// buffering so prefetch overlaps WMMA compute with no VGPR round-trip.
// ---------------------------------------------------------------------------

typedef __bf16 bf16_t;
typedef __attribute__((ext_vector_type(16))) __bf16 v16bf;
typedef __attribute__((ext_vector_type(8)))  float  v8f;

#define SEQ 2048
#define EMB 1024
#define NH  16
#define HD  64
#define MTOT 4096   // B*S

union Frag16 { v16bf v; unsigned int u[8]; };

__device__ __forceinline__ bf16_t tobf(float x) { return (bf16_t)x; }

// CDNA5 async copy: 16 bytes per lane, global memory -> LDS, no VGPR data.
// VDST VGPR holds the wave-relative LDS byte offset; per the ISA aperture
// rules the low 32 bits of a flat shared-pointer are exactly that offset.
__device__ __forceinline__ void async_ld_b128(void* lds_dst, const void* gsrc) {
  const unsigned loff = (unsigned)(unsigned long long)lds_dst;
  asm volatile("global_load_async_to_lds_b128 %0, %1, off"
               :: "v"(loff), "v"(gsrc)
               : "memory");
}

__device__ __forceinline__ void wait_async0() {
  asm volatile("s_wait_asynccnt 0x0" ::: "memory");
}

// A-matrix fragment: 16x32 bf16, LDS tile is [row][k] with row-stride `ld`.
// lanes 0-15 row=lane, VGPR i holds K={2i,2i+1};
// lanes 16-31 same rows, K offset +16.  Pairs are K-contiguous -> b32 loads
// (the compiler merges the consecutive ones into ds_load_b128).
__device__ __forceinline__ v16bf load_a_frag(const bf16_t* __restrict__ tile,
                                             int mBase, int kBase, int ld) {
  const int lane = threadIdx.x & 31;
  const int hi   = lane >> 4;
  const int row  = mBase + (lane & 15);
  Frag16 f;
#pragma unroll
  for (int i = 0; i < 8; ++i) {
    const int k = kBase + 2 * i + (hi ? 16 : 0);
    f.u[i] = *(const unsigned int*)(tile + row * ld + k);
  }
  return f.v;
}

// B-matrix fragment: 32(K)x16(N) bf16, LDS tile stored TRANSPOSED as [n][k]
// with row-stride `ld`.  VGPR v lo/hi = K={2v,2v+1} (+16 for lanes 16-31),
// N = lane&15.  [n][k] layout makes each VGPR one contiguous b32 load.
__device__ __forceinline__ v16bf load_b_frag(const bf16_t* __restrict__ tileT,
                                             int nBase, int kBase, int ld) {
  const int lane = threadIdx.x & 31;
  const int hi   = lane >> 4;
  const int n    = nBase + (lane & 15);
  Frag16 f;
#pragma unroll
  for (int i = 0; i < 8; ++i) {
    const int k = kBase + 2 * i + (hi ? 16 : 0);
    f.u[i] = *(const unsigned int*)(tileT + n * ld + k);
  }
  return f.v;
}

__device__ __forceinline__ v8f wmma_bf16(v16bf a, v16bf b, v8f c) {
  return __builtin_amdgcn_wmma_f32_16x16x32_bf16(
      /*neg_a=*/false, a, /*neg_b=*/false, b,
      /*c_mod=*/(short)0, c, /*reuse_a=*/false, /*reuse_b=*/false);
}

// ---------------------------------------------------------------------------
// Prep kernels: fp32 -> bf16 conversion, and weight transpose+convert so
// GEMM B-tiles are already [n][k] in global memory.
// ---------------------------------------------------------------------------
__global__ __launch_bounds__(256) void cvt_f32_bf16_kernel(
    const float* __restrict__ src, bf16_t* __restrict__ dst, int n) {
  const int i = (blockIdx.x * 256 + threadIdx.x) * 4;
  if (i + 3 < n) {
    const float4 v = *(const float4*)(src + i);
    dst[i + 0] = tobf(v.x);
    dst[i + 1] = tobf(v.y);
    dst[i + 2] = tobf(v.z);
    dst[i + 3] = tobf(v.w);
  }
}

__global__ __launch_bounds__(256) void transpose_cvt_kernel(
    const float* __restrict__ W, bf16_t* __restrict__ WT, int K, int N) {
  const int idx = blockIdx.x * 256 + threadIdx.x;
  if (idx >= K * N) return;
  const int k = idx / N;
  const int n = idx - k * N;
  WT[(size_t)n * K + k] = tobf(W[idx]);
}

// ---------------------------------------------------------------------------
// QKV projection: C[4096][3072] = X[4096][1024] @ Wqkv + b, then scatter to
// Q [bh][s][d] (pre-scaled by 1/sqrt(HD)), K [bh][s][d], V^T [bh][d][s].
// Block tile 128x128, 8 waves (2x4), each wave 64x32 = 4x2 WMMA tiles.
// Async double-buffered staging: prefetch k-tile kt+32 while WMMAing kt.
// ---------------------------------------------------------------------------
__global__ __launch_bounds__(256) void qkv_gemm_kernel(
    const bf16_t* __restrict__ X,     // [4096][1024]
    const bf16_t* __restrict__ WT,    // [3072][1024]  ([n][k])
    const float*  __restrict__ bias,  // [3072]
    bf16_t* __restrict__ Qw,          // [32][2048][64]
    bf16_t* __restrict__ Kw,          // [32][2048][64]
    bf16_t* __restrict__ Vtw) {       // [32][64][2048]
  __shared__ __align__(16) bf16_t As[2][128 * 32];
  __shared__ __align__(16) bf16_t Bs[2][128 * 32];

  const int tid   = threadIdx.x;
  const int lane  = tid & 31;
  const int w     = tid >> 5;
  const int hi    = lane >> 4;
  const int waveM = w >> 2;   // 0..1
  const int waveN = w & 3;    // 0..3
  const int blockM = blockIdx.y * 128;
  const int blockN = blockIdx.x * 128;

  v8f acc[4][2];
#pragma unroll
  for (int i = 0; i < 4; ++i) { acc[i][0] = {}; acc[i][1] = {}; }

  const int ldRow = tid >> 1;        // 0..127
  const int ldCol = (tid & 1) * 16;  // 0 or 16

  auto stage = [&](int kt, int buf) {
    {  // A tile [128 m][32 k]
      const bf16_t* src = X + (size_t)(blockM + ldRow) * EMB + kt + ldCol;
      bf16_t* dst = As[buf] + ldRow * 32 + ldCol;
      async_ld_b128(dst, src);
      async_ld_b128(dst + 8, src + 8);
    }
    {  // B tile [128 n][32 k]
      const bf16_t* src = WT + (size_t)(blockN + ldRow) * EMB + kt + ldCol;
      bf16_t* dst = Bs[buf] + ldRow * 32 + ldCol;
      async_ld_b128(dst, src);
      async_ld_b128(dst + 8, src + 8);
    }
  };

  stage(0, 0);
  int jj = 0;
  for (int kt = 0; kt < EMB; kt += 32, jj ^= 1) {
    wait_async0();      // this wave's async writes complete
    __syncthreads();    // all waves' writes visible
    if (kt + 32 < EMB) stage(kt + 32, jj ^ 1);  // prefetch next tile

    const v16bf bfr0 = load_b_frag(Bs[jj], waveN * 32 + 0, 0, 32);
    const v16bf bfr1 = load_b_frag(Bs[jj], waveN * 32 + 16, 0, 32);
#pragma unroll
    for (int ms = 0; ms < 4; ++ms) {
      const v16bf af = load_a_frag(As[jj], waveM * 64 + ms * 16, 0, 32);
      acc[ms][0] = wmma_bf16(af, bfr0, acc[ms][0]);
      acc[ms][1] = wmma_bf16(af, bfr1, acc[ms][1]);
    }
  }

  // Epilogue: C layout = VGPR r -> row r (lanes 0-15) / r+8 (lanes 16-31),
  // lane&15 -> col.  Scatter into Q / K / V^T head layouts.
#pragma unroll
  for (int ns = 0; ns < 2; ++ns) {
    const int nG = blockN + waveN * 32 + ns * 16 + (lane & 15);
    const int region = nG >> 10;  // 0=Q 1=K 2=V
    const int e = nG & 1023;
    const int h = e >> 6;
    const int d = e & 63;
    const float bv = bias[nG];
#pragma unroll
    for (int ms = 0; ms < 4; ++ms) {
#pragma unroll
      for (int r = 0; r < 8; ++r) {
        const int mG = blockM + waveM * 64 + ms * 16 + r + hi * 8;
        const int bb = mG >> 11;    // batch
        const int s  = mG & 2047;
        const int bh = bb * NH + h;
        const float v = acc[ms][ns][r] + bv;
        if (region == 0)
          Qw[((size_t)bh * SEQ + s) * HD + d] = tobf(v * 0.125f);  // 1/sqrt(64)
        else if (region == 1)
          Kw[((size_t)bh * SEQ + s) * HD + d] = tobf(v);
        else
          Vtw[((size_t)bh * HD + d) * SEQ + s] = tobf(v);
      }
    }
  }
}

// ---------------------------------------------------------------------------
// Flash attention: one block = (bh, 128 q rows), 8 waves x 16 q rows.
// Stream keys 32 at a time: QK^T (4 WMMA) -> online softmax -> P via LDS
// bounce -> PV (4 WMMA).  K/V tiles double-buffered via async-to-LDS so the
// next key block streams in during softmax + PV.  Scores never touch HBM.
// ---------------------------------------------------------------------------
__global__ __launch_bounds__(256) void attention_kernel(
    const bf16_t* __restrict__ Qw,   // [32][2048][64], pre-scaled
    const bf16_t* __restrict__ Kw,   // [32][2048][64]
    const bf16_t* __restrict__ Vtw,  // [32][64][2048]
    bf16_t* __restrict__ ctx) {      // [4096][1024]
  __shared__ __align__(16) bf16_t Qs[128 * 64];      // 16 KB
  __shared__ __align__(16) bf16_t Ks[2][32 * 64];    //  8 KB ([key][d])
  __shared__ __align__(16) bf16_t Vts[2][64 * 32];   //  8 KB ([d][key])
  __shared__ __align__(16) bf16_t Ps[8 * 16 * 32];   //  8 KB wave-private P

  const int tid  = threadIdx.x;
  const int lane = tid & 31;
  const int w    = tid >> 5;
  const int hi   = lane >> 4;
  const int bh   = blockIdx.y;        // 0..31
  const int qBase = blockIdx.x * 128;

  const bf16_t* Qg = Qw + (size_t)bh * SEQ * HD;
  const bf16_t* Kg = Kw + (size_t)bh * SEQ * HD;
  const bf16_t* Vg = Vtw + (size_t)bh * HD * SEQ;

  auto stageKV = [&](int j, int buf) {
    {  // K tile [32 keys][64 d]
      const int row = tid >> 3;       // 0..31
      const int col = (tid & 7) * 8;  // 0..56
      async_ld_b128(Ks[buf] + row * 64 + col,
                    Kg + (size_t)(j + row) * HD + col);
    }
    {  // V^T tile [64 d][32 keys]
      const int row = tid >> 2;       // 0..63
      const int col = (tid & 3) * 8;  // 0..24
      async_ld_b128(Vts[buf] + row * 32 + col,
                    Vg + (size_t)row * SEQ + j + col);
    }
  };

  {  // stage Q tile [128][64] once (async as well)
    const int row = tid >> 1;
    const int col = (tid & 1) * 32;
    const bf16_t* src = Qg + (size_t)(qBase + row) * HD + col;
    bf16_t* dst = Qs + row * 64 + col;
#pragma unroll
    for (int i = 0; i < 4; ++i)
      async_ld_b128(dst + i * 8, src + i * 8);
  }
  stageKV(0, 0);
  wait_async0();
  __syncthreads();

  const v16bf qa0 = load_a_frag(Qs, w * 16, 0, 64);
  const v16bf qa1 = load_a_frag(Qs, w * 16, 32, 64);

  v8f O[4];
#pragma unroll
  for (int i = 0; i < 4; ++i) O[i] = {};
  float mrow[8], lrow[8];
#pragma unroll
  for (int r = 0; r < 8; ++r) { mrow[r] = -1e30f; lrow[r] = 0.f; }

  bf16_t* Pw = Ps + w * (16 * 32);

  int jj = 0;
  for (int j = 0; j < SEQ; j += 32, jj ^= 1) {
    wait_async0();
    __syncthreads();
    if (j + 32 < SEQ) stageKV(j + 32, jj ^ 1);  // prefetch next key block

    // S = Q(16x64) @ K^T(64x32): two 16x16 n-tiles, two k-steps each.
    v8f s0 = {}, s1 = {};
    {
      v16bf kb;
      kb = load_b_frag(Ks[jj], 0, 0, 64);   s0 = wmma_bf16(qa0, kb, s0);
      kb = load_b_frag(Ks[jj], 0, 32, 64);  s0 = wmma_bf16(qa1, kb, s0);
      kb = load_b_frag(Ks[jj], 16, 0, 64);  s1 = wmma_bf16(qa0, kb, s1);
      kb = load_b_frag(Ks[jj], 16, 32, 64); s1 = wmma_bf16(qa1, kb, s1);
    }

    // Online softmax.  Row r lives in VGPR r across a 16-lane half; reduce
    // across lanes with xor-shuffles that stay inside the half (masks < 16).
#pragma unroll
    for (int r = 0; r < 8; ++r) {
      float mt = fmaxf(s0[r], s1[r]);
#pragma unroll
      for (int off = 1; off < 16; off <<= 1)
        mt = fmaxf(mt, __shfl_xor(mt, off, 32));
      const float mn    = fmaxf(mrow[r], mt);
      const float alpha = __expf(mrow[r] - mn);
      const float p0 = __expf(s0[r] - mn);
      const float p1 = __expf(s1[r] - mn);
      float rs = p0 + p1;
#pragma unroll
      for (int off = 1; off < 16; off <<= 1)
        rs += __shfl_xor(rs, off, 32);
      lrow[r] = lrow[r] * alpha + rs;
      mrow[r] = mn;
#pragma unroll
      for (int t = 0; t < 4; ++t) O[t][r] *= alpha;
      // C-layout -> A-layout via wave-private LDS (in-order DS per wave).
      const int prow = r + hi * 8;
      Pw[prow * 32 + (lane & 15)]      = tobf(p0);
      Pw[prow * 32 + 16 + (lane & 15)] = tobf(p1);
    }

    // O += P(16x32) @ V(32x64)
    const v16bf pa = load_a_frag(Pw, 0, 0, 32);
#pragma unroll
    for (int t = 0; t < 4; ++t)
      O[t] = wmma_bf16(pa, load_b_frag(Vts[jj], t * 16, 0, 32), O[t]);
  }

  // Epilogue: normalize by running sum, store context as [b][s][h*64+d] bf16.
  const int bb = bh >> 4;
  const int h  = bh & 15;
  float inv[8];
#pragma unroll
  for (int r = 0; r < 8; ++r) inv[r] = 1.0f / lrow[r];
#pragma unroll
  for (int t = 0; t < 4; ++t) {
    const int d = t * 16 + (lane & 15);
#pragma unroll
    for (int r = 0; r < 8; ++r) {
      const int row = qBase + w * 16 + r + hi * 8;
      ctx[((size_t)(bb * SEQ + row)) * EMB + h * HD + d] = tobf(O[t][r] * inv[r]);
    }
  }
}

// ---------------------------------------------------------------------------
// Output projection: out[4096][1024] = ctx @ Wout + b_out (fp32 result).
// ---------------------------------------------------------------------------
__global__ __launch_bounds__(256) void out_gemm_kernel(
    const bf16_t* __restrict__ A,    // ctx [4096][1024]
    const bf16_t* __restrict__ WT,   // [1024][1024] ([n][k])
    const float*  __restrict__ bias, // [1024]
    float* __restrict__ out) {       // [4096][1024]
  __shared__ __align__(16) bf16_t As[2][128 * 32];
  __shared__ __align__(16) bf16_t Bs[2][128 * 32];

  const int tid   = threadIdx.x;
  const int lane  = tid & 31;
  const int w     = tid >> 5;
  const int hi    = lane >> 4;
  const int waveM = w >> 2;
  const int waveN = w & 3;
  const int blockM = blockIdx.y * 128;
  const int blockN = blockIdx.x * 128;

  v8f acc[4][2];
#pragma unroll
  for (int i = 0; i < 4; ++i) { acc[i][0] = {}; acc[i][1] = {}; }

  const int ldRow = tid >> 1;
  const int ldCol = (tid & 1) * 16;

  auto stage = [&](int kt, int buf) {
    {
      const bf16_t* src = A + (size_t)(blockM + ldRow) * EMB + kt + ldCol;
      bf16_t* dst = As[buf] + ldRow * 32 + ldCol;
      async_ld_b128(dst, src);
      async_ld_b128(dst + 8, src + 8);
    }
    {
      const bf16_t* src = WT + (size_t)(blockN + ldRow) * EMB + kt + ldCol;
      bf16_t* dst = Bs[buf] + ldRow * 32 + ldCol;
      async_ld_b128(dst, src);
      async_ld_b128(dst + 8, src + 8);
    }
  };

  stage(0, 0);
  int jj = 0;
  for (int kt = 0; kt < EMB; kt += 32, jj ^= 1) {
    wait_async0();
    __syncthreads();
    if (kt + 32 < EMB) stage(kt + 32, jj ^ 1);

    const v16bf bfr0 = load_b_frag(Bs[jj], waveN * 32 + 0, 0, 32);
    const v16bf bfr1 = load_b_frag(Bs[jj], waveN * 32 + 16, 0, 32);
#pragma unroll
    for (int ms = 0; ms < 4; ++ms) {
      const v16bf af = load_a_frag(As[jj], waveM * 64 + ms * 16, 0, 32);
      acc[ms][0] = wmma_bf16(af, bfr0, acc[ms][0]);
      acc[ms][1] = wmma_bf16(af, bfr1, acc[ms][1]);
    }
  }

#pragma unroll
  for (int ns = 0; ns < 2; ++ns) {
    const int nG = blockN + waveN * 32 + ns * 16 + (lane & 15);
    const float bv = bias[nG];
#pragma unroll
    for (int ms = 0; ms < 4; ++ms) {
#pragma unroll
      for (int r = 0; r < 8; ++r) {
        const int mG = blockM + waveM * 64 + ms * 16 + r + hi * 8;
        out[(size_t)mG * EMB + nG] = acc[ms][ns][r] + bv;
      }
    }
  }
}

// ---------------------------------------------------------------------------
// Launch.  Workspace layout (48 MB total):
//   [ 0,  8) MB  x bf16            [4096][1024]
//   [ 8, 14) MB  Wqkv^T bf16       [3072][1024]
//   [14, 16) MB  Wout^T bf16       [1024][1024]
//   [16, 24) MB  Q bf16 (scaled)   [32][2048][64]
//   [24, 32) MB  K bf16            [32][2048][64]
//   [32, 40) MB  V^T bf16          [32][64][2048]
//   [40, 48) MB  ctx bf16          [4096][1024]
// ---------------------------------------------------------------------------
extern "C" void kernel_launch(void* const* d_in, const int* in_sizes, int n_in,
                              void* d_out, int out_size, void* d_ws, size_t ws_size,
                              hipStream_t stream) {
  (void)in_sizes; (void)n_in; (void)out_size; (void)ws_size;
  const float* x     = (const float*)d_in[0];
  const float* W_qkv = (const float*)d_in[1];
  const float* b_qkv = (const float*)d_in[2];
  const float* W_out = (const float*)d_in[3];
  const float* b_out = (const float*)d_in[4];
  float* out = (float*)d_out;

  char* ws = (char*)d_ws;
  bf16_t* xb    = (bf16_t*)(ws);
  bf16_t* WqkvT = (bf16_t*)(ws + ((size_t)8  << 20));
  bf16_t* WoutT = (bf16_t*)(ws + ((size_t)14 << 20));
  bf16_t* Qw    = (bf16_t*)(ws + ((size_t)16 << 20));
  bf16_t* Kw    = (bf16_t*)(ws + ((size_t)24 << 20));
  bf16_t* Vtw   = (bf16_t*)(ws + ((size_t)32 << 20));
  bf16_t* ctx   = (bf16_t*)(ws + ((size_t)40 << 20));

  // prep
  cvt_f32_bf16_kernel<<<(MTOT * EMB) / (4 * 256), 256, 0, stream>>>(x, xb, MTOT * EMB);
  transpose_cvt_kernel<<<(EMB * 3 * EMB) / 256, 256, 0, stream>>>(W_qkv, WqkvT, EMB, 3 * EMB);
  transpose_cvt_kernel<<<(EMB * EMB) / 256, 256, 0, stream>>>(W_out, WoutT, EMB, EMB);

  // QKV projection (M=4096, N=3072, K=1024)
  qkv_gemm_kernel<<<dim3(3 * EMB / 128, MTOT / 128), 256, 0, stream>>>(
      xb, WqkvT, b_qkv, Qw, Kw, Vtw);

  // flash attention (16 q-tiles x 32 batch*head)
  attention_kernel<<<dim3(SEQ / 128, 2 * NH), 256, 0, stream>>>(Qw, Kw, Vtw, ctx);

  // output projection (M=4096, N=1024, K=1024)
  out_gemm_kernel<<<dim3(EMB / 128, MTOT / 128), 256, 0, stream>>>(
      ctx, WoutT, b_out, out);
}
